// GRUWithAttentionModel_64381559767202
// MI455X (gfx1250) — matrix operand order repaired and checked by reading, not compiled, not run on hardware
//
#include <hip/hip_runtime.h>
#include <hip/hip_bf16.h>
#include <stdint.h>

// ---------------------------------------------------------------------------
// GRUWithAttentionModel on MI455X (gfx1250): bf16 WMMA pipeline, v3.
// B=4, D=60, S=512, F=158, C=64, H=128, OUT=1
// v3: explicit addrspace(1) for hot global traffic (global_* not flat_*),
//     no pointer-array selects, GRU split to 128 WGs (16 seqs each).
// ---------------------------------------------------------------------------

#define Bb 4
#define Dd 60
#define Ss 512
#define Ff 158
#define Cc 64
#define Hh 128
#define KF 160        // F padded to mult of 32
#define KA 224        // F+C=222 padded to mult of 32
#define NN (Bb*Ss)    // 2048 GRU sequences
#define BD (Bb*Dd)    // 240

// ---- workspace layout (byte offsets, all 16B aligned) ----
#define WS_WQT  0u                      // bf16 [64][160]  (c-major, zero-pad K)
#define WS_WKT  20480u
#define WS_WVT  40960u
#define WS_WIH  61440u                  // bf16 [384][224] (row n contiguous in k)
#define WS_WHH  233472u                 // bf16 [384][128]
#define WS_Q    331776u                 // bf16 [240][512][64]
#define WS_K    16060416u
#define WS_V    31789056u
#define WS_MKT  47517696u               // f32  [240][64]
#define WS_AUG  47579136u               // bf16 [60][2048][224]
// end: 102629376 bytes

typedef float  v8f  __attribute__((ext_vector_type(8)));
typedef __bf16 v8bf __attribute__((ext_vector_type(8)));
typedef __bf16 v16bf __attribute__((ext_vector_type(16)));
typedef unsigned int u32x4 __attribute__((ext_vector_type(4)));
typedef int i32x8 __attribute__((ext_vector_type(8)));
typedef int i32x4 __attribute__((ext_vector_type(4)));
typedef unsigned int u32x4v __attribute__((ext_vector_type(4)));   // 16B copy unit

// ---- explicit global (addrspace 1) pointers: emit global_*, not flat_* ----
#define GAS __attribute__((address_space(1)))
typedef GAS const v8bf*   gc_v8bf;
typedef GAS const float*  gc_f32;
typedef GAS float*        g_f32;
typedef GAS const __bf16* gc_bf;
typedef GAS __bf16*       g_bf;
typedef GAS const u32x4v* gc_u4;
typedef GAS const int*    gc_i32;
static __device__ inline unsigned long long uptr(const void* p) {
  return (unsigned long long)(uintptr_t)p;
}

#if defined(__has_builtin)
#  if __has_builtin(__builtin_amdgcn_tensor_load_to_lds) && __has_builtin(__builtin_amdgcn_s_wait_tensorcnt)
#    define HAVE_TDM 1
#  endif
#endif
#ifndef HAVE_TDM
#  define HAVE_TDM 0
#endif

// ---- WMMA fragment helpers (layouts per CDNA5 ISA 7.12.2, wave32) ----
#define CAT16(lo, hi) __builtin_shufflevector(lo, hi, 0,1,2,3,4,5,6,7,8,9,10,11,12,13,14,15)

// A 16x32 bf16 from LDS: lane m=lane&15; lanes<16 K 0..7 & 16..23, lanes>=16 K 8..15 & 24..31
static __device__ inline v16bf afrag(const __bf16* base, int stride, int kc, int lane) {
  int m  = lane & 15;
  int kb = (lane < 16) ? 0 : 8;
  const __bf16* p = base + m * stride + kc * 32 + kb;
  v8bf lo = *(const v8bf*)p;
  v8bf hi = *(const v8bf*)(p + 16);
  return CAT16(lo, hi);
}
static __device__ inline v16bf afrag_g(const __bf16* base, int stride, int kc, int lane) {
  int m  = lane & 15;
  int kb = (lane < 16) ? 0 : 8;
  unsigned long long p = uptr(base) + 2ull * ((size_t)m * stride + kc * 32 + kb);
  v8bf lo = *(gc_v8bf)p;
  v8bf hi = *(gc_v8bf)(p + 32);
  return CAT16(lo, hi);
}
// B 32x16 bf16 from "n-major" buffer bt[n][k]: lane n=lane&15; lanes<16 K 0..7 & 16..23, lanes>=16 K 8..15 & 24..31
static __device__ inline v16bf bfrag(const __bf16* bt, int strideN, int kc, int lane) {
  int n  = lane & 15;
  int kb = (lane < 16) ? 0 : 8;
  const __bf16* p = bt + n * strideN + kc * 32 + kb;
  v8bf lo = *(const v8bf*)p;
  v8bf hi = *(const v8bf*)(p + 16);
  return CAT16(lo, hi);
}
static __device__ inline v16bf bfrag_g(const __bf16* bt, int strideN, int kc, int lane) {
  int n  = lane & 15;
  int kb = (lane < 16) ? 0 : 8;
  unsigned long long p = uptr(bt) + 2ull * ((size_t)n * strideN + kc * 32 + kb);
  v8bf lo = *(gc_v8bf)p;
  v8bf hi = *(gc_v8bf)(p + 32);
  return CAT16(lo, hi);
}
static __device__ inline v8f wmma_bf16(v16bf a, v16bf b, v8f c) {
  return __builtin_amdgcn_wmma_f32_16x16x32_bf16(false, a, false, b, (short)0, c, false, false);
}
static __device__ inline float hred_max(float v) {
  v = fmaxf(v, __shfl_xor(v, 1, 32)); v = fmaxf(v, __shfl_xor(v, 2, 32));
  v = fmaxf(v, __shfl_xor(v, 4, 32)); v = fmaxf(v, __shfl_xor(v, 8, 32));
  return v;
}
static __device__ inline float hred_add(float v) {
  v += __shfl_xor(v, 1, 32); v += __shfl_xor(v, 2, 32);
  v += __shfl_xor(v, 4, 32); v += __shfl_xor(v, 8, 32);
  return v;
}
static __device__ inline float wred_add(float v) {
  v = hred_add(v); v += __shfl_xor(v, 16, 32); return v;
}
static __device__ inline float sigmoidf(float x) { return 1.f / (1.f + __expf(-x)); }

// ---------------------------------------------------------------------------
// Kernel 1: weight conversion fp32 -> bf16 with padding / layout swap
// ---------------------------------------------------------------------------
__global__ void k_prep(const float* __restrict__ wq, const float* __restrict__ wk,
                       const float* __restrict__ wv, const float* __restrict__ w_ih,
                       const float* __restrict__ w_hh, char* __restrict__ ws) {
  int i = blockIdx.x * 256 + threadIdx.x;
  const int QKV = 3 * Cc * KF;          // 30720
  const int IH  = 384 * KA;             // 86016
  const int HHN = 384 * Hh;             // 49152
  if (i < QKV) {
    int r = i / (Cc * KF), j = i % (Cc * KF);
    int c = j / KF, k = j % KF;
    const float* w = (r == 0) ? wq : (r == 1) ? wk : wv;
    unsigned off = (r == 0) ? WS_WQT : (r == 1) ? WS_WKT : WS_WVT;
    float v = (k < Ff) ? *(gc_f32)(uptr(w) + 4ull * (k * Cc + c)) : 0.f;
    *(g_bf)(uptr(ws) + off + 2ull * (c * KF + k)) = (__bf16)v;
  } else if ((i -= QKV) < IH) {
    int n = i / KA, k = i % KA;
    float v = (k < Ff + Cc) ? *(gc_f32)(uptr(w_ih) + 4ull * (n * (Ff + Cc) + k)) : 0.f;
    *(g_bf)(uptr(ws) + WS_WIH + 2ull * (n * KA + k)) = (__bf16)v;
  } else if ((i -= IH) < HHN) {
    int n = i / Hh, k = i % Hh;
    float v = *(gc_f32)(uptr(w_hh) + 4ull * (n * Hh + k));
    *(g_bf)(uptr(ws) + WS_WHH + 2ull * (n * Hh + k)) = (__bf16)v;
  }
}

// ---------------------------------------------------------------------------
// Kernel 2: Q/K/V projection.  grid = BD*(S/128), 256 thr (8 waves).
// ---------------------------------------------------------------------------
static __device__ __forceinline__ void proj_one(const __bf16* Xs, int m0, int lane, int srow0,
                                                const __bf16* wt, const float* bias,
                                                __bf16* outp) {
  for (int j = 0; j < 4; ++j) {
    v8f acc = {};
#pragma unroll
    for (int kc = 0; kc < 5; ++kc) {
      v16bf a = afrag(Xs + m0 * KF, KF, kc, lane);
      v16bf b = bfrag_g(wt + (j * 16) * KF, KF, kc, lane);
      acc = wmma_bf16(a, b, acc);
    }
    int c = j * 16 + (lane & 15);
    float bb = *(gc_f32)(uptr(bias) + 4ull * c);
    int rbase = srow0 + m0 + ((lane < 16) ? 0 : 8);
#pragma unroll
    for (int ii = 0; ii < 8; ++ii)
      *(g_bf)(uptr(outp) + 2ull * ((size_t)(rbase + ii) * Cc + c)) = (__bf16)(acc[ii] + bb);
  }
}

__global__ void k_qkv(const float* __restrict__ x, const float* __restrict__ bq,
                      const float* __restrict__ bk, const float* __restrict__ bv,
                      char* __restrict__ ws) {
  __shared__ __bf16 Xs[128 * KF];       // 40 KB
  int bd = blockIdx.x >> 2;
  int srow0 = (blockIdx.x & 3) * 128;
  unsigned long long xg = uptr(x) + 4ull * (size_t)(bd * Ss + srow0) * Ff;
  for (int i = threadIdx.x; i < 128 * KF; i += 256) {
    int r = i / KF, k = i % KF;
    Xs[i] = (k < Ff) ? (__bf16)(*(gc_f32)(xg + 4ull * (r * Ff + k))) : (__bf16)0.f;
  }
  __syncthreads();
  int lane = threadIdx.x & 31, w = threadIdx.x >> 5;
  int m0 = w * 16;
  size_t tileoff = (size_t)bd * Ss * Cc;
  proj_one(Xs, m0, lane, srow0, (const __bf16*)(ws + WS_WQT), bq, (__bf16*)(ws + WS_Q) + tileoff);
  proj_one(Xs, m0, lane, srow0, (const __bf16*)(ws + WS_WKT), bk, (__bf16*)(ws + WS_K) + tileoff);
  proj_one(Xs, m0, lane, srow0, (const __bf16*)(ws + WS_WVT), bv, (__bf16*)(ws + WS_V) + tileoff);
}

// ---------------------------------------------------------------------------
// Kernel 3: masked attention + masked mean -> market[bd][64].
// grid = BD, 256 thr.  Flash-style online softmax, 32 keys / iteration.
// ---------------------------------------------------------------------------
__global__ void k_attn(const int* __restrict__ mask, char* __restrict__ ws) {
  extern __shared__ char smem[];
  __bf16* Ks    = (__bf16*)smem;                 // [512][64]   64 KB
  __bf16* Vt    = (__bf16*)(smem + 65536);       // [64][512]   64 KB (transposed)
  __bf16* Ps    = (__bf16*)(smem + 131072);      // 8 waves x 16x32 scratch, 8 KB
  float*  biasL = (float*)(smem + 139264);       // [512] key bias 0 / -1e9
  float*  mkt   = (float*)(smem + 141312);       // [64]
  int*    dcnt  = (int*)(smem + 141568);

  int bd = blockIdx.x;
  int tid = threadIdx.x, lane = tid & 31, w = tid >> 5;
  if (tid < 64) mkt[tid] = 0.f;
  if (tid == 0) *dcnt = 0;
  __syncthreads();

  unsigned long long Kg = uptr(ws) + WS_K + 2ull * (size_t)bd * Ss * Cc;
  unsigned long long Vg = uptr(ws) + WS_V + 2ull * (size_t)bd * Ss * Cc;
  const __bf16* Qg = (const __bf16*)(ws + WS_Q) + (size_t)bd * Ss * Cc;
  { // stage K (b128 copies), V transposed, key-bias + denom
    u32x4v* dst = (u32x4v*)Ks;
    for (int i = tid; i < (Ss * Cc * 2) / 16; i += 256) dst[i] = *(gc_u4)(Kg + 16ull * i);
    for (int i = tid; i < Ss * Cc; i += 256) {
      int s = i >> 6, c = i & 63;
      Vt[c * Ss + s] = *(gc_bf)(Vg + 2ull * i);
    }
    for (int i = tid; i < Ss; i += 256) {
      int mv = *(gc_i32)(uptr(mask) + 4ull * ((size_t)bd * Ss + i));
      biasL[i] = mv ? 0.f : -1e9f;
      if (mv) atomicAdd(dcnt, 1);
    }
  }
  __syncthreads();

  __bf16* myP = Ps + w * 512;           // 16x32 per-wave scratch
  for (int t = 0; t < 4; ++t) {
    int q0 = (w * 4 + t) * 16;
    v16bf aq0 = afrag_g(Qg + q0 * Cc, Cc, 0, lane);
    v16bf aq1 = afrag_g(Qg + q0 * Cc, Cc, 1, lane);
    v8f acc0 = {}, acc1 = {}, acc2 = {}, acc3 = {};
    float mrow[8], lrow[8];
#pragma unroll
    for (int i = 0; i < 8; ++i) { mrow[i] = -3.0e38f; lrow[i] = 0.f; }

    for (int kb = 0; kb < Ss; kb += 32) {
      v8f s0 = {}, s1 = {};
      s0 = wmma_bf16(aq0, bfrag(Ks + (size_t)kb * Cc, Cc, 0, lane), s0);
      s0 = wmma_bf16(aq1, bfrag(Ks + (size_t)kb * Cc, Cc, 1, lane), s0);
      s1 = wmma_bf16(aq0, bfrag(Ks + (size_t)(kb + 16) * Cc, Cc, 0, lane), s1);
      s1 = wmma_bf16(aq1, bfrag(Ks + (size_t)(kb + 16) * Cc, Cc, 1, lane), s1);
      float kb0 = biasL[kb + (lane & 15)];
      float kb1 = biasL[kb + 16 + (lane & 15)];
      float p0[8], p1[8];
#pragma unroll
      for (int i = 0; i < 8; ++i) {
        float a0 = s0[i] * 0.125f + kb0;   // 1/sqrt(64)
        float a1 = s1[i] * 0.125f + kb1;
        float tm = hred_max(fmaxf(a0, a1));
        float mnew = fmaxf(mrow[i], tm);
        float scale = __expf(mrow[i] - mnew);
        float e0 = __expf(a0 - mnew);
        float e1 = __expf(a1 - mnew);
        lrow[i] = lrow[i] * scale + hred_add(e0 + e1);
        mrow[i] = mnew;
        acc0[i] *= scale; acc1[i] *= scale; acc2[i] *= scale; acc3[i] *= scale;
        p0[i] = e0; p1[i] = e1;
      }
      int hlf = (lane < 16) ? 0 : 8;
#pragma unroll
      for (int i = 0; i < 8; ++i) {        // D-tile -> A-fragment reshape via LDS
        myP[(i + hlf) * 32 + (lane & 15)] = (__bf16)p0[i];
        myP[(i + hlf) * 32 + 16 + (lane & 15)] = (__bf16)p1[i];
      }
      v16bf ap = afrag(myP, 32, 0, lane);
      acc0 = wmma_bf16(ap, bfrag(Vt + 0 * 16 * Ss + kb, Ss, 0, lane), acc0);
      acc1 = wmma_bf16(ap, bfrag(Vt + 1 * 16 * Ss + kb, Ss, 0, lane), acc1);
      acc2 = wmma_bf16(ap, bfrag(Vt + 2 * 16 * Ss + kb, Ss, 0, lane), acc2);
      acc3 = wmma_bf16(ap, bfrag(Vt + 3 * 16 * Ss + kb, Ss, 0, lane), acc3);
    }
    int hlf = (lane < 16) ? 0 : 8;
    float part0 = 0.f, part1 = 0.f, part2 = 0.f, part3 = 0.f;
#pragma unroll
    for (int i = 0; i < 8; ++i) {
      float qm = (biasL[q0 + i + hlf] == 0.f) ? 1.f : 0.f;
      float inv = qm / lrow[i];
      part0 += acc0[i] * inv; part1 += acc1[i] * inv;
      part2 += acc2[i] * inv; part3 += acc3[i] * inv;
    }
    part0 += __shfl_xor(part0, 16, 32); part1 += __shfl_xor(part1, 16, 32);
    part2 += __shfl_xor(part2, 16, 32); part3 += __shfl_xor(part3, 16, 32);
    if (lane < 16) {
      atomicAdd(&mkt[ 0 + lane], part0);
      atomicAdd(&mkt[16 + lane], part1);
      atomicAdd(&mkt[32 + lane], part2);
      atomicAdd(&mkt[48 + lane], part3);
    }
  }
  __syncthreads();
  if (tid < 64) {
    float dn = fmaxf((float)*dcnt, 1.f);
    *(g_f32)(uptr(ws) + WS_MKT + 4ull * ((size_t)bd * Cc + tid)) = mkt[tid] / dn;
  }
}

// ---------------------------------------------------------------------------
// Kernel 4: concat + layernorm -> bf16 aug[d][n][224].  One wave per row.
// ---------------------------------------------------------------------------
__global__ void k_augln(const float* __restrict__ x, const float* __restrict__ ln_g,
                        const float* __restrict__ ln_b, char* __restrict__ ws) {
  int row = blockIdx.x * 8 + (threadIdx.x >> 5);
  int lane = threadIdx.x & 31;
  int bd = row >> 9, s = row & 511;
  int b = bd / Dd, d = bd % Dd;
  unsigned long long mk = uptr(ws) + WS_MKT + 4ull * (size_t)bd * Cc;
  unsigned long long xr = uptr(x) + 4ull * (size_t)row * Ff;
  float vals[7], sum = 0.f, sq = 0.f;
#pragma unroll
  for (int ii = 0; ii < 7; ++ii) {
    int e = lane + ii * 32;
    float v = 0.f;
    if (e < Ff) v = *(gc_f32)(xr + 4ull * e);
    else if (e < Ff + Cc) v = *(gc_f32)(mk + 4ull * (e - Ff));
    vals[ii] = v;
    if (e < Ff + Cc) { sum += v; sq += v * v; }
  }
  sum = wred_add(sum); sq = wred_add(sq);
  const float invn = 1.f / (float)(Ff + Cc);
  float mean = sum * invn;
  float var = sq * invn - mean * mean;
  float inv = rsqrtf(var + 1e-5f);
  int n = b * Ss + s;
  unsigned long long out = uptr(ws) + WS_AUG + 2ull * ((size_t)d * NN + n) * KA;
#pragma unroll
  for (int ii = 0; ii < 7; ++ii) {
    int e = lane + ii * 32;
    float v = (e < Ff + Cc) ? (vals[ii] - mean) * inv * ln_g[e] + ln_b[e] : 0.f;
    *(g_bf)(out + 2ull * e) = (__bf16)v;
  }
}

// ---------------------------------------------------------------------------
// Kernel 5: GRU scan over 60 days + MLP head. 128 WGs x 16 sequences.
// Per step each wave owns one 16-col gate tile j = waveId:
// 3 gates x 7 kc (input proj) + 3 gates x 4 kc (recurrent) = 33 WMMA.
// aug tile fetched with the Tensor Data Mover when available.
// LDS: whh 96KB @0, aug 7KB @98304, h x2 4KB @105472/@109568 -> 113664 B.
// ---------------------------------------------------------------------------
__global__ void k_gru(const float* __restrict__ b_ih, const float* __restrict__ b_hh,
                      const float* __restrict__ w1, const float* __restrict__ b1,
                      const float* __restrict__ w2, const float* __restrict__ b2,
                      const char* __restrict__ ws, float* __restrict__ out) {
  extern __shared__ char smem[];
  __bf16* whh = (__bf16*)smem;                  // [384][128]  96 KB
  __bf16* aug = (__bf16*)(smem + 98304);        // [16][224]    7 KB
  __bf16* hb0 = (__bf16*)(smem + 105472);       // [16][128]    4 KB
  __bf16* hb1 = (__bf16*)(smem + 109568);       // [16][128]    4 KB
  int tid = threadIdx.x, lane = tid & 31, w = tid >> 5;
  { // stage w_hh (n-major bf16) + zero h
    unsigned long long src = uptr(ws) + WS_WHH;
    u32x4v* dst = (u32x4v*)whh;
    for (int i = tid; i < (384 * Hh * 2) / 16; i += 256) dst[i] = *(gc_u4)(src + 16ull * i);
    for (int i = tid; i < 16 * Hh; i += 256) hb0[i] = (__bf16)0.f;
  }
  __syncthreads();

  int n0 = blockIdx.x * 16;
  __bf16* hc = hb0; __bf16* hn = hb1;
  int j = w;                                     // gate tile owned by this wave
  const __bf16* wih = (const __bf16*)(ws + WS_WIH);

  for (int d = 0; d < Dd; ++d) {
    const __bf16* asrc = (const __bf16*)(ws + WS_AUG) + ((size_t)d * NN + n0) * KA;
#if HAVE_TDM
    if (w == 0) {   // one wave issues the TDM DMA: 16x224 bf16 tile -> LDS @98304
      unsigned long long ga = uptr(asrc);
      u32x4 g0 = {};
      g0[0] = 1u;                                  // count=1 valid descriptor
      g0[1] = 98304u;                              // lds_addr (bytes)
      g0[2] = (unsigned)(ga & 0xffffffffu);        // global_addr lo
      g0[3] = (unsigned)((ga >> 32) & 0x1ffffffu) | 0x80000000u;  // addr hi | type=2
      i32x8 g1 = {};
      g1[0] = (int)(1u << 16);                     // data_size = 2 bytes
      g1[1] = (int)((unsigned)KA << 16);           // tensor_dim0 = 224
      g1[2] = (int)((unsigned)NN << 16);           // tensor_dim1 = 2048
      g1[3] = (int)((unsigned)KA << 16);           // tile_dim0 = 224
      g1[4] = 16;                                  // tile_dim1 = 16
      g1[5] = KA;                                  // tensor_dim0_stride = 224
      i32x4 z4 = {};
      i32x8 z8 = {};
      __builtin_amdgcn_tensor_load_to_lds(g0, g1, z4, z4, z8, 0);
      __builtin_amdgcn_s_wait_tensorcnt(0);
    }
#else
    {
      unsigned long long a4 = uptr(asrc);
      u32x4v* dst = (u32x4v*)aug;
      for (int i = tid; i < (16 * KA * 2) / 16; i += 256) dst[i] = *(gc_u4)(a4 + 16ull * i);
    }
#endif
    if (d + 1 < Dd) __builtin_prefetch(asrc + (size_t)NN * KA, 0, 1);
    __syncthreads();

    {
      v8f xr = {}, xz = {}, xn = {}, hr = {}, hz = {}, hv = {};
#pragma unroll
      for (int kc = 0; kc < 7; ++kc) {
        v16bf a = afrag(aug, KA, kc, lane);
        xr = wmma_bf16(a, bfrag_g(wih + (size_t)(0   + j * 16) * KA, KA, kc, lane), xr);
        xz = wmma_bf16(a, bfrag_g(wih + (size_t)(128 + j * 16) * KA, KA, kc, lane), xz);
        xn = wmma_bf16(a, bfrag_g(wih + (size_t)(256 + j * 16) * KA, KA, kc, lane), xn);
      }
#pragma unroll
      for (int kc = 0; kc < 4; ++kc) {
        v16bf a = afrag(hc, Hh, kc, lane);
        hr = wmma_bf16(a, bfrag(whh + (0   + j * 16) * Hh, Hh, kc, lane), hr);
        hz = wmma_bf16(a, bfrag(whh + (128 + j * 16) * Hh, Hh, kc, lane), hz);
        hv = wmma_bf16(a, bfrag(whh + (256 + j * 16) * Hh, Hh, kc, lane), hv);
      }
      int c = j * 16 + (lane & 15);
      float bir = *(gc_f32)(uptr(b_ih) + 4ull * c);
      float biz = *(gc_f32)(uptr(b_ih) + 4ull * (128 + c));
      float bin = *(gc_f32)(uptr(b_ih) + 4ull * (256 + c));
      float bhr = *(gc_f32)(uptr(b_hh) + 4ull * c);
      float bhz = *(gc_f32)(uptr(b_hh) + 4ull * (128 + c));
      float bhn = *(gc_f32)(uptr(b_hh) + 4ull * (256 + c));
      int hlf = (lane < 16) ? 0 : 8;
#pragma unroll
      for (int i = 0; i < 8; ++i) {
        int r = i + hlf;
        float rr = sigmoidf(xr[i] + bir + hr[i] + bhr);
        float zz = sigmoidf(xz[i] + biz + hz[i] + bhz);
        float nv = tanhf(xn[i] + bin + rr * (hv[i] + bhn));
        float ho = (float)hc[r * Hh + c];
        hn[r * Hh + c] = (__bf16)((1.f - zz) * nv + zz * ho);
      }
    }
    __bf16* tp = hc; hc = hn; hn = tp;
    __syncthreads();
  }
  // --- MLP head: relu(h @ w1 + b1) @ w2 + b2 ---
  if (tid < 16) {
    const __bf16* hrow = hc + tid * Hh;
    float o = b2[0];
    for (int k = 0; k < Hh / 2; ++k) {
      float acc = b1[k];
      for (int h = 0; h < Hh; ++h)
        acc += (float)hrow[h] * (*(gc_f32)(uptr(w1) + 4ull * (h * (Hh / 2) + k)));
      o += fmaxf(acc, 0.f) * w2[k];
    }
    *(g_f32)(uptr(out) + 4ull * (n0 + tid)) = o;
  }
}

// ---------------------------------------------------------------------------
extern "C" void kernel_launch(void* const* d_in, const int* in_sizes, int n_in,
                              void* d_out, int out_size, void* d_ws, size_t ws_size,
                              hipStream_t stream) {
  (void)in_sizes; (void)n_in; (void)out_size; (void)ws_size;
  const float* x    = (const float*)d_in[0];
  const int*   mask = (const int*)d_in[1];
  const float* wq   = (const float*)d_in[2];
  const float* bq   = (const float*)d_in[3];
  const float* wk   = (const float*)d_in[4];
  const float* bk   = (const float*)d_in[5];
  const float* wv   = (const float*)d_in[6];
  const float* bv   = (const float*)d_in[7];
  const float* ln_g = (const float*)d_in[8];
  const float* ln_b = (const float*)d_in[9];
  const float* w_ih = (const float*)d_in[10];
  const float* w_hh = (const float*)d_in[11];
  const float* b_ih = (const float*)d_in[12];
  const float* b_hh = (const float*)d_in[13];
  const float* w1   = (const float*)d_in[14];
  const float* b1   = (const float*)d_in[15];
  const float* w2   = (const float*)d_in[16];
  const float* b2   = (const float*)d_in[17];
  char*  ws  = (char*)d_ws;
  float* out = (float*)d_out;

  k_prep <<<648, 256, 0, stream>>>(wq, wk, wv, w_ih, w_hh, ws);
  k_qkv  <<<BD * (Ss / 128), 256, 0, stream>>>(x, bq, bk, bv, ws);
  k_attn <<<BD, 256, 141824, stream>>>(mask, ws);
  k_augln<<<(Bb * Dd * Ss) / 8, 256, 0, stream>>>(x, ln_g, ln_b, ws);
  k_gru  <<<NN / 16, 256, 113664, stream>>>(b_ih, b_hh, w1, b1, w2, b2, ws, out);
}
